// Attention_6064493822346
// MI455X (gfx1250) — compile-verified, matrix operand order checked by reading, not run on hardware
//
#include <hip/hip_runtime.h>
#include <hip/hip_bf16.h>

// ---------------------------------------------------------------------------
// Types for CDNA5 WMMA (wave32): A/B = v16h (16 halves = 8 VGPRs),
// C/D = v8f (8 floats = 8 VGPRs).
// ---------------------------------------------------------------------------
typedef __attribute__((ext_vector_type(16))) _Float16 v16h;
typedef __attribute__((ext_vector_type(8)))  _Float16 v8h;
typedef __attribute__((ext_vector_type(8)))  float    v8f;
typedef int v4i __attribute__((vector_size(16)));   // matches async-LDS builtin

#define NSEQ   4096
#define DMODEL 1024
#define HEADS  16
#define DHEAD  64
#define INNER  (HEADS * DHEAD)   // 1024
#define QKVW   (3 * INNER)       // 3072

// ---------------------------------------------------------------------------
// gfx1250 async global->LDS copy (ASYNCcnt-tracked). Probe-confirmed builtin
// signature: (AS1 int4* src, AS3 int4* dst, imm offset, imm cpol).
// Guarded so the file compiles on toolchains without the builtin.
// ---------------------------------------------------------------------------
#if __has_builtin(__builtin_amdgcn_global_load_async_to_lds_b128)
#define USE_ASYNC_LDS 1
#endif

__device__ __forceinline__ void stage16(_Float16* l, const _Float16* g) {
#ifdef USE_ASYNC_LDS
  __builtin_amdgcn_global_load_async_to_lds_b128(
      (__attribute__((address_space(1))) v4i*)(_Float16*)g,
      (__attribute__((address_space(3))) v4i*)l, 0, 0);
#else
  *(v8h*)l = *(const v8h*)g;
#endif
}

__device__ __forceinline__ void stage_wait() {
#ifdef USE_ASYNC_LDS
#if __has_builtin(__builtin_amdgcn_s_wait_asynccnt)
  __builtin_amdgcn_s_wait_asynccnt(0);
#else
  asm volatile("s_wait_asynccnt 0" ::: "memory");
#endif
#endif
}

// ---------------------------------------------------------------------------
// Fragment loader following the documented CDNA5 16-bit A-matrix 16x32 layout:
//   lane l (0..31): row = row0 + (l & 15)
//   K chunk base   = k0 + ((l & 16) ? 8 : 0)
//   halves 0..7  <- K = base .. base+7      (contiguous 16 bytes)
//   halves 8..15 <- K = base+16 .. base+23  (contiguous 16 bytes)
// The B operand (K x 16) is loaded with the same pattern from rows of B^T,
// which by symmetry of the striping gives lane = column N, halves = K.
// Works for global or LDS-backed pointers (inlined -> addrspace inferred).
// ---------------------------------------------------------------------------
__device__ __forceinline__ v16h load_frag(const _Float16* __restrict__ p,
                                          int ld, int row0, int k0) {
  const int lane = threadIdx.x & 31;
  const _Float16* a =
      p + (size_t)(row0 + (lane & 15)) * (size_t)ld + (k0 + ((lane & 16) ? 8 : 0));
  v8h lo = *(const v8h*)(a);
  v8h hi = *(const v8h*)(a + 16);
  v16h f;
#pragma unroll
  for (int i = 0; i < 8; ++i) { f[i] = lo[i]; f[i + 8] = hi[i]; }
  return f;
}

__device__ __forceinline__ v8f wmma_f16(v16h a, v16h b, v8f c) {
  // (neg_a, A, neg_b, B, c_mod, C, reuse_a, reuse_b)
  return __builtin_amdgcn_wmma_f32_16x16x32_f16(false, a, false, b, (short)0, c,
                                                false, false);
}

// ---------------------------------------------------------------------------
// 1) LayerNorm: one workgroup per row of x, output f16 activation.
// ---------------------------------------------------------------------------
__global__ void __launch_bounds__(256) ln_kernel(const float* __restrict__ x,
                                                 const float* __restrict__ g,
                                                 const float* __restrict__ b,
                                                 _Float16* __restrict__ h16) {
  __shared__ float red[256];
  const int row = blockIdx.x;
  const float* xr = x + (size_t)row * DMODEL;
  float s = 0.f, ss = 0.f;
  for (int c = threadIdx.x; c < DMODEL; c += 256) {
    float v = xr[c];
    s += v; ss += v * v;
  }
  red[threadIdx.x] = s; __syncthreads();
  for (int o = 128; o > 0; o >>= 1) {
    if (threadIdx.x < o) red[threadIdx.x] += red[threadIdx.x + o];
    __syncthreads();
  }
  const float mean = red[0] * (1.0f / DMODEL);
  __syncthreads();
  red[threadIdx.x] = ss; __syncthreads();
  for (int o = 128; o > 0; o >>= 1) {
    if (threadIdx.x < o) red[threadIdx.x] += red[threadIdx.x + o];
    __syncthreads();
  }
  const float var  = red[0] * (1.0f / DMODEL) - mean * mean;
  const float rstd = rsqrtf(var + 1e-5f);
  for (int c = threadIdx.x; c < DMODEL; c += 256)
    h16[(size_t)row * DMODEL + c] =
        (_Float16)((xr[c] - mean) * rstd * g[c] + b[c]);
}

// ---------------------------------------------------------------------------
// 2) Cast f32 weight [K][N] -> f16 transposed [N][K] (so GEMM B-operand loads
//    are contiguous row-pattern reads of B^T).
// ---------------------------------------------------------------------------
__global__ void __launch_bounds__(256) cast_transpose_kernel(
    const float* __restrict__ W, _Float16* __restrict__ WT, int K, int Ncols) {
  size_t tid = (size_t)blockIdx.x * blockDim.x + threadIdx.x;
  if (tid >= (size_t)K * Ncols) return;
  int k = (int)(tid / Ncols);
  int n = (int)(tid % Ncols);
  WT[(size_t)n * K + k] = (_Float16)W[tid];
}

// ---------------------------------------------------------------------------
// 3) WMMA GEMM: C[M,N](f32) = A16[M,K] * BT16[N,K]^T.
//    128 threads = 4 waves arranged 2x2; workgroup tile 128x128 staged through
//    LDS per K-step of 32 (async global->LDS when available); each wave owns a
//    64x64 register tile (16 f32 accumulators, 16 v_wmma per K-step).
//    Global intensity ~64 FLOP/B; LDS intensity ~32 FLOP/B -> WMMA-limited.
// ---------------------------------------------------------------------------
#define BM 128
#define BN 128
#define BK 32

__global__ void __launch_bounds__(128) gemm_wmma_kernel(
    const _Float16* __restrict__ A, const _Float16* __restrict__ BT,
    float* __restrict__ C, int M, int N, int K) {
  __shared__ __align__(16) _Float16 ldsA[BM * BK];
  __shared__ __align__(16) _Float16 ldsB[BN * BK];

  const int tid    = threadIdx.x;
  const int waveId = tid >> 5;
  const int lane   = tid & 31;
  const int wm0 = (waveId & 1) * 64;   // wave origin inside the block tile
  const int wn0 = (waveId >> 1) * 64;
  const int bm = blockIdx.x * BM;
  const int bn = blockIdx.y * BN;

  v8f acc[4][4] = {};

  for (int k = 0; k < K; k += BK) {
    if (k + BK < K) {  // emits global_prefetch_b8 for the next K-tile
      __builtin_prefetch(A + (size_t)bm * K + k + BK, 0, 3);
      __builtin_prefetch(BT + (size_t)bn * K + k + BK, 0, 3);
    }
    // Stage A and B tiles (128 rows x 32 halves = 8KB each) into LDS.
    // 512 16-byte chunks per tile; 128 threads x 4 chunks.
#pragma unroll
    for (int i = 0; i < 4; ++i) {
      const int c   = tid + i * 128;
      const int row = c >> 2;
      const int col = (c & 3) * 8;               // halves
      stage16(ldsA + row * BK + col, A + (size_t)(bm + row) * K + k + col);
      stage16(ldsB + row * BK + col, BT + (size_t)(bn + row) * K + k + col);
    }
    stage_wait();          // ASYNCcnt==0 for this wave's copies
    __syncthreads();       // all waves' tiles visible

    v16h af[4];
#pragma unroll
    for (int mt = 0; mt < 4; ++mt)
      af[mt] = load_frag(ldsA, BK, wm0 + mt * 16, 0);
#pragma unroll
    for (int nt = 0; nt < 4; ++nt) {
      v16h bf = load_frag(ldsB, BK, wn0 + nt * 16, 0);
#pragma unroll
      for (int mt = 0; mt < 4; ++mt)
        acc[mt][nt] = wmma_f16(af[mt], bf, acc[mt][nt]);
    }
    __syncthreads();       // done reading LDS before next stage overwrites
  }

  // C/D layout: lane -> N = lane&15 ; VGPR r -> M = r + ((lane&16)?8:0)
  const int col  = lane & 15;
  const int roff = (lane & 16) ? 8 : 0;
#pragma unroll
  for (int mt = 0; mt < 4; ++mt)
#pragma unroll
    for (int nt = 0; nt < 4; ++nt)
#pragma unroll
      for (int r = 0; r < 8; ++r)
        C[(size_t)(bm + wm0 + mt * 16 + r + roff) * N +
          (bn + wn0 + nt * 16 + col)] = acc[mt][nt][r];
}

// ---------------------------------------------------------------------------
// 4) RoPE + split: read fused qkv f32 [n][3072];
//    - q,k: interleaved-pair RoPE -> f16 head-major [h][n][64]
//    - v: f16 transposed [h][64][n] (so PV's B-operand is row-pattern loads)
//    - new_k/new_v: pre-RoPE f32 straight into d_out ([h][n][64])
//    One thread per (n, h, pair p of 2 elements): N*H*32 threads.
// ---------------------------------------------------------------------------
__global__ void __launch_bounds__(256) rope_split_kernel(
    const float* __restrict__ qkv, _Float16* __restrict__ q16,
    _Float16* __restrict__ k16, _Float16* __restrict__ vT16,
    float* __restrict__ nk, float* __restrict__ nv) {
  int tid = blockIdx.x * blockDim.x + threadIdx.x;
  int n   = tid >> 9;        // / (HEADS*32)
  int rem = tid & 511;
  int h   = rem >> 5;
  int p   = rem & 31;

  const float* row = qkv + (size_t)n * QKVW;
  const int c = h * DHEAD + 2 * p;
  float qx = row[c],            qy = row[c + 1];
  float kx = row[INNER + c],    ky = row[INNER + c + 1];
  float vx = row[2*INNER + c],  vy = row[2*INNER + c + 1];

  // theta^(-2p/64); ln(10000) = 9.210340371976184
  float invf = __expf(-(float)(2 * p) * (1.0f / DHEAD) * 9.210340371976184f);
  float ang  = (float)n * invf;
  float cs = __cosf(ang), sn = __sinf(ang);

  size_t hb = ((size_t)h * NSEQ + n) * DHEAD;
  q16[hb + 2*p]     = (_Float16)(qx * cs - qy * sn);
  q16[hb + 2*p + 1] = (_Float16)(qy * cs + qx * sn);
  k16[hb + 2*p]     = (_Float16)(kx * cs - ky * sn);
  k16[hb + 2*p + 1] = (_Float16)(ky * cs + kx * sn);

  nk[hb + 2*p] = kx; nk[hb + 2*p + 1] = ky;   // cache holds pre-RoPE k
  nv[hb + 2*p] = vx; nv[hb + 2*p + 1] = vy;

  size_t vb = (size_t)h * DHEAD * NSEQ;
  vT16[vb + (size_t)(2*p) * NSEQ + n]     = (_Float16)vx;
  vT16[vb + (size_t)(2*p + 1) * NSEQ + n] = (_Float16)vy;
}

// ---------------------------------------------------------------------------
// 5) Causal flash attention, 1 wave per (16 q-rows, head).
//    Per 32-key block: 4 WMMAs for S, online softmax (16-lane shfl_xor row
//    reductions in C/D layout), P -> LDS f16 (C-layout -> A-layout transpose,
//    guarded by s_wait_dscnt), 4 WMMAs for O against V^T.
// ---------------------------------------------------------------------------
__global__ void __launch_bounds__(32) attn_kernel(
    const _Float16* __restrict__ q16, const _Float16* __restrict__ k16,
    const _Float16* __restrict__ vT16, _Float16* __restrict__ o16) {
  __shared__ __align__(16) _Float16 plds[16 * 32];

  const int lane = threadIdx.x & 31;
  const int qi0  = blockIdx.x * 16;
  const int h    = blockIdx.y;

  const _Float16* qh = q16 + (size_t)h * NSEQ * DHEAD;
  const _Float16* kh = k16 + (size_t)h * NSEQ * DHEAD;
  const _Float16* vh = vT16 + (size_t)h * DHEAD * NSEQ;

  const v16h qf0 = load_frag(qh, DHEAD, qi0, 0);
  const v16h qf1 = load_frag(qh, DHEAD, qi0, 32);

  v8f o0 = {}, o1 = {}, o2 = {}, o3 = {};
  float mrow[8], lrow[8];
#pragma unroll
  for (int r = 0; r < 8; ++r) { mrow[r] = -3.0e38f; lrow[r] = 0.f; }

  const int roff = (lane & 16) ? 8 : 0;   // C/D row offset for this half-wave
  const int colN = lane & 15;             // C/D column (key index in tile)
  const float scale = 0.125f;             // 1/sqrt(64)

  for (int kb = 0; kb <= qi0; kb += 32) {
    // S = Q * K^T for keys [kb, kb+32): two 16-col tiles, K-dim = DHEAD
    v8f s0 = {}, s1 = {};
    s0 = wmma_f16(qf0, load_frag(kh, DHEAD, kb, 0),       s0);
    s0 = wmma_f16(qf1, load_frag(kh, DHEAD, kb, 32),      s0);
    s1 = wmma_f16(qf0, load_frag(kh, DHEAD, kb + 16, 0),  s1);
    s1 = wmma_f16(qf1, load_frag(kh, DHEAD, kb + 16, 32), s1);

#pragma unroll
    for (int r = 0; r < 8; ++r) {
      const int i = qi0 + r + roff;            // query row of this accum slot
      float v0 = s0[r] * scale; if (kb + colN > i)      v0 = -3.0e38f;
      float v1 = s1[r] * scale; if (kb + 16 + colN > i) v1 = -3.0e38f;

      float rmax = fmaxf(v0, v1);
#pragma unroll
      for (int off = 1; off < 16; off <<= 1)
        rmax = fmaxf(rmax, __shfl_xor(rmax, off, 16));
      const float mn   = fmaxf(mrow[r], rmax);
      const float corr = __expf(mrow[r] - mn);
      const float p0 = __expf(v0 - mn);
      const float p1 = __expf(v1 - mn);
      float rs = p0 + p1;
#pragma unroll
      for (int off = 1; off < 16; off <<= 1)
        rs += __shfl_xor(rs, off, 16);
      lrow[r] = lrow[r] * corr + rs;
      mrow[r] = mn;
      o0[r] *= corr; o1[r] *= corr; o2[r] *= corr; o3[r] *= corr;

      // C-layout -> row-major 16x32 f16 tile in LDS
      plds[(r + roff) * 32 + colN]      = (_Float16)p0;
      plds[(r + roff) * 32 + 16 + colN] = (_Float16)p1;
    }
    // LDS is in-order per wave; make the A-fragment reads wait for stores.
    asm volatile("s_wait_dscnt 0" ::: "memory");

    const v16h pf = load_frag(plds, 32, 0, 0);   // A-layout P (16x32)
    o0 = wmma_f16(pf, load_frag(vh, NSEQ, 0,  kb), o0);
    o1 = wmma_f16(pf, load_frag(vh, NSEQ, 16, kb), o1);
    o2 = wmma_f16(pf, load_frag(vh, NSEQ, 32, kb), o2);
    o3 = wmma_f16(pf, load_frag(vh, NSEQ, 48, kb), o3);
  }

  // Normalize and scatter to merged-head layout [n][h*64] (f16, GEMM A input)
#pragma unroll
  for (int r = 0; r < 8; ++r) {
    const float inv = 1.0f / lrow[r];
    const size_t base = (size_t)(qi0 + r + roff) * INNER + h * DHEAD;
    o16[base + colN]      = (_Float16)(o0[r] * inv);
    o16[base + 16 + colN] = (_Float16)(o1[r] * inv);
    o16[base + 32 + colN] = (_Float16)(o2[r] * inv);
    o16[base + 48 + colN] = (_Float16)(o3[r] * inv);
  }
}

// ---------------------------------------------------------------------------
// Host-side orchestration (graph-capture safe: only kernel launches on stream)
// ---------------------------------------------------------------------------
extern "C" void kernel_launch(void* const* d_in, const int* in_sizes, int n_in,
                              void* d_out, int out_size, void* d_ws,
                              size_t ws_size, hipStream_t stream) {
  (void)in_sizes; (void)n_in; (void)out_size; (void)ws_size;

  const float* x     = (const float*)d_in[0];
  const float* w_qkv = (const float*)d_in[1];
  const float* w_out = (const float*)d_in[2];
  const float* ln_g  = (const float*)d_in[3];
  const float* ln_b  = (const float*)d_in[4];

  float* out   = (float*)d_out;                             // [4096*1024]
  float* new_k = out + (size_t)NSEQ * DMODEL;               // [16][4096][64]
  float* new_v = new_k + (size_t)HEADS * NSEQ * DHEAD;      // [16][4096][64]

  char* ws = (char*)d_ws;
  _Float16* h16    = (_Float16*)(ws);                       //  8 MB
  _Float16* wqkvT  = (_Float16*)(ws + 8388608);             //  6 MB
  _Float16* woutT  = (_Float16*)(ws + 14680064);            //  2 MB
  float*    qkv32  = (float*)   (ws + 16777216);            // 48 MB
  _Float16* q16    = (_Float16*)(ws + 67108864);            //  8 MB
  _Float16* k16    = (_Float16*)(ws + 75497472);            //  8 MB
  _Float16* vT16   = (_Float16*)(ws + 83886080);            //  8 MB
  _Float16* attn16 = (_Float16*)(ws + 92274688);            //  8 MB

  // 1) LayerNorm -> f16
  ln_kernel<<<NSEQ, 256, 0, stream>>>(x, ln_g, ln_b, h16);

  // 2) Weight casts (transposed)
  cast_transpose_kernel<<<(DMODEL * QKVW + 255) / 256, 256, 0, stream>>>(
      w_qkv, wqkvT, DMODEL, QKVW);
  cast_transpose_kernel<<<(INNER * DMODEL + 255) / 256, 256, 0, stream>>>(
      w_out, woutT, INNER, DMODEL);

  // 3) QKV GEMM: [4096,1024] x [1024,3072] -> f32 [4096,3072]
  gemm_wmma_kernel<<<dim3(NSEQ / BM, QKVW / BN), 128, 0, stream>>>(
      h16, wqkvT, qkv32, NSEQ, QKVW, DMODEL);

  // 4) RoPE + split + KV-cache outputs
  rope_split_kernel<<<(NSEQ * HEADS * 32 + 255) / 256, 256, 0, stream>>>(
      qkv32, q16, k16, vT16, new_k, new_v);

  // 5) Causal flash attention (1 wave per 16 query rows per head)
  attn_kernel<<<dim3(NSEQ / 16, HEADS), 32, 0, stream>>>(q16, k16, vT16,
                                                         attn16);

  // 6) Output projection: [4096,1024] x [1024,1024] -> f32 out
  gemm_wmma_kernel<<<dim3(NSEQ / BM, DMODEL / BN), 128, 0, stream>>>(
      attn16, woutT, out, NSEQ, DMODEL, DMODEL);
}
